// HashFilter_13022340842287
// MI455X (gfx1250) — matrix operand order repaired and checked by reading, not compiled, not run on hardware
//
#include <hip/hip_runtime.h>
#include <hip/hip_bf16.h>

typedef _Float16 v16h __attribute__((ext_vector_type(16)));
typedef _Float16 v8h  __attribute__((ext_vector_type(8)));
typedef float    v8f  __attribute__((ext_vector_type(8)));

#define BATCH 8
#define TABLE_NUM 16
#define TABLE_DIM 8192
#define RESO 256
#define CH 32
#define STYLE_DIM 512
#define PAD 258          // 256 + 1-pixel halo each side

__device__ __constant__ int RES_LIST_C[16] =
    {16,19,23,27,33,40,48,58,70,84,101,122,147,176,212,256};

__device__ __forceinline__ void corner1d(float gc, int res, int& c0, int& c1, float& f){
    float pos = gc * (float)(res - 1);
    float p0f = floorf(pos);
    f = pos - p0f;
    int p0 = (int)p0f;
    if (p0 < 0) p0 = 0;
    if (p0 > res - 2) p0 = res - 2;
    c0 = p0; c1 = p0 + 1;
}

__device__ __forceinline__ int table_idx(int cx, int cy, int res){
    if (res * res <= TABLE_DIM) return cy * res + cx;
    unsigned h = (unsigned)cx ^ ((unsigned)cy * 2654435761u);
    return (int)(h & (unsigned)(TABLE_DIM - 1));
}

// ---------------- zero scratch ----------------
__global__ void zero_kernel(float* p, int n){
    int i = blockIdx.x * blockDim.x + threadIdx.x;
    if (i < n) p[i] = 0.0f;
}

// ---------------- styles + demod coefficients ----------------
__global__ void styles_kernel(const float* __restrict__ s,
                              const float* __restrict__ affine_w,
                              const float* __restrict__ affine_b,
                              const float* __restrict__ conv_w,
                              float* __restrict__ styles,
                              float* __restrict__ dcoef){
    int b = blockIdx.x;
    int c = threadIdx.x;           // 32 threads = one wave
    __shared__ float st[CH];
    float acc = 0.0f;
    for (int k = 0; k < STYLE_DIM; k++)
        acc += s[b * STYLE_DIM + k] * affine_w[c * STYLE_DIM + k];
    float sv = acc * 0.04419417382415922f /* 1/sqrt(512) */ + affine_b[c];
    st[c] = sv;
    styles[b * CH + c] = sv;
    __syncthreads();
    float d = 0.0f;
    for (int i = 0; i < CH; i++){
        const float* wp = conv_w + ((size_t)c * CH + i) * 9;
        float wsq = 0.0f;
        #pragma unroll
        for (int t = 0; t < 9; t++){ float w = wp[t]; wsq += w * w; }
        d += wsq * st[i] * st[i];
    }
    dcoef[b * CH + c] = rsqrtf(d + 1e-8f);
}

// ---------------- weight repack: conv_w[o][i][ky][kx] -> f16 wt[tap][o][i] ----------------
__global__ void wprep_kernel(const float* __restrict__ conv_w, _Float16* __restrict__ wt){
    int idx = blockIdx.x * blockDim.x + threadIdx.x;     // 9*32*32 = 9216
    if (idx >= 9 * CH * CH) return;
    int t = idx / (CH * CH);
    int n = (idx % (CH * CH)) / CH;
    int i = idx % CH;
    wt[idx] = (_Float16)conv_w[((size_t)n * CH + i) * 9 + t];
}

// ---------------- hash-grid gather + style modulation -> f16 NHWC with halo ----------------
__global__ void retrieve_kernel(const float* __restrict__ tables,
                                const float* __restrict__ styles,
                                _Float16* __restrict__ xm){
    int b = blockIdx.y;
    int pix = blockIdx.x * blockDim.x + threadIdx.x;
    if (pix >= PAD * PAD) return;
    int py = pix / PAD, px = pix % PAD;
    _Float16* dst = xm + ((size_t)(b * PAD + py) * PAD + px) * CH;
    if (py == 0 || py == PAD - 1 || px == 0 || px == PAD - 1){
        v8h z = {};
        #pragma unroll
        for (int q = 0; q < 4; q++) ((v8h*)dst)[q] = z;
        return;
    }
    int i = py - 1, j = px - 1;                    // (y, x)
    float gx = (float)j * (1.0f / 255.0f);
    float gy = (float)i * (1.0f / 255.0f);
    __align__(16) _Float16 row[CH];
    #pragma unroll 4
    for (int l = 0; l < TABLE_NUM; l++){
        int res = RES_LIST_C[l];
        int x0, x1, y0, y1; float fx, fy;
        corner1d(gx, res, x0, x1, fx);
        corner1d(gy, res, y0, y1, fy);
        float w00 = (1.0f - fx) * (1.0f - fy), w10 = fx * (1.0f - fy);
        float w01 = (1.0f - fx) * fy,          w11 = fx * fy;
        const float2* tab = ((const float2*)tables) + (size_t)(b * TABLE_NUM + l) * TABLE_DIM;
        float2 g00 = tab[table_idx(x0, y0, res)];
        float2 g10 = tab[table_idx(x1, y0, res)];
        float2 g01 = tab[table_idx(x0, y1, res)];
        float2 g11 = tab[table_idx(x1, y1, res)];
        float f0 = w00 * g00.x + w10 * g10.x + w01 * g01.x + w11 * g11.x;
        float f1 = w00 * g00.y + w10 * g10.y + w01 * g01.y + w11 * g11.y;
        row[2 * l]     = (_Float16)(f0 * styles[b * CH + 2 * l]);
        row[2 * l + 1] = (_Float16)(f1 * styles[b * CH + 2 * l + 1]);
    }
    #pragma unroll
    for (int q = 0; q < 4; q++) ((v8h*)dst)[q] = *(const v8h*)(row + 8 * q);
}

// ---------------- WMMA implicit-GEMM conv + demod/bias/lrelu + fused scatter ----------------
__global__ __launch_bounds__(128) void conv_kernel(
        const _Float16* __restrict__ xm, const _Float16* __restrict__ wt,
        const float* __restrict__ dcoef, const float* __restrict__ conv_b,
        float* __restrict__ recon){
    __shared__ _Float16 swt[9 * CH * CH];          // 18 KB weight tile
    int tid = threadIdx.x;
    {   // cooperative weight stage (4608 dwords)
        const uint32_t* src = (const uint32_t*)wt;
        uint32_t* d = (uint32_t*)swt;
        for (int k = tid; k < 9 * CH * CH / 2; k += 128) d[k] = src[k];
    }
    __syncthreads();

    int lane = tid & 31, wave = tid >> 5;
    int b = blockIdx.z, y = blockIdx.y;
    int x0 = (blockIdx.x * 4 + wave) * 16;
    int m  = lane & 15;
    int K0 = (lane >= 16) ? 8  : 0;   // A-fragment K base (ISA 16-bit A 16x32 layout)
    int Kb = (lane >= 16) ? 8 : 0;    // B-fragment K base
    int n0 = lane & 15;               // B-fragment column

    v8f acc0 = {}; v8f acc1 = {};

    // Single per-lane base pointer; all 9 taps are compile-time byte offsets
    // (dy*PAD*CH + dx*CH halves, max ~33 KB -> fits the signed 24-bit VMEM imm).
    const _Float16* ap0 = xm
        + ((size_t)b * PAD * PAD + (size_t)y * PAD + (size_t)(x0 + m)) * CH + K0;
    const _Float16* bp0 = swt + n0 * CH + Kb;              // +1024*t per tap (imm)
    const _Float16* bp1 = swt + (n0 + 16) * CH + Kb;

    #pragma unroll
    for (int dy = 0; dy < 3; dy++){
        #pragma unroll
        for (int dx = 0; dx < 3; dx++){
            const int aoff = (dy * PAD + dx) * CH;         // constant after unroll
            v8h alo = *(const v8h*)(ap0 + aoff);
            v8h ahi = *(const v8h*)(ap0 + aoff + 16);
            v16h A;
            #pragma unroll
            for (int r = 0; r < 8; r++){ A[r] = alo[r]; A[8 + r] = ahi[r]; }

            const int t = dy * 3 + dx;
            const int woff = t * CH * CH;                  // constant after unroll
            v8h b0a = *(const v8h*)(bp0 + woff), b0b = *(const v8h*)(bp0 + woff + 16);
            v8h b1a = *(const v8h*)(bp1 + woff), b1b = *(const v8h*)(bp1 + woff + 16);
            v16h B0, B1;
            #pragma unroll
            for (int r = 0; r < 8; r++){
                B0[r] = b0a[r]; B0[8 + r] = b0b[r];
                B1[r] = b1a[r]; B1[8 + r] = b1b[r];
            }
            acc0 = __builtin_amdgcn_wmma_f32_16x16x32_f16(false, A, false, B0, (short)0, acc0, false, false);
            acc1 = __builtin_amdgcn_wmma_f32_16x16x32_f16(false, A, false, B1, (short)0, acc1, false, false);
        }
    }

    // Epilogue: demod + bias + lrelu*sqrt2, then transpose scatter-add into tables.
    float gy = (float)y * (1.0f / 255.0f);
    int mbase = (lane >= 16) ? 8 : 0;             // C/D layout: VGPR r -> M=r (+8 for upper lanes)
    #pragma unroll
    for (int nt = 0; nt < 2; nt++){
        v8f acc = nt ? acc1 : acc0;
        int ch = n0 + 16 * nt;
        float dc = dcoef[b * CH + ch];
        float cb = conv_b[ch];
        int l = ch >> 1, f = ch & 1;
        int res = RES_LIST_C[l];
        int y0c, y1c; float fy;
        corner1d(gy, res, y0c, y1c, fy);
        float* rb = recon + ((size_t)(b * TABLE_NUM + l) * TABLE_DIM) * 2 + f;
        #pragma unroll
        for (int r = 0; r < 8; r++){
            int p = x0 + r + mbase;               // pixel x
            float val = acc[r] * dc + cb;
            val = (val < 0.0f ? 0.2f * val : val) * 1.4142135623730951f;
            float gx = (float)p * (1.0f / 255.0f);
            int x0c, x1c; float fx;
            corner1d(gx, res, x0c, x1c, fx);
            float w00 = (1.0f - fx) * (1.0f - fy), w10 = fx * (1.0f - fy);
            float w01 = (1.0f - fx) * fy,          w11 = fx * fy;
            atomicAdd(rb + 2 * table_idx(x0c, y0c, res), val * w00);
            atomicAdd(rb + 2 * table_idx(x1c, y0c, res), val * w10);
            atomicAdd(rb + 2 * table_idx(x0c, y1c, res), val * w01);
            atomicAdd(rb + 2 * table_idx(x1c, y1c, res), val * w11);
        }
    }
}

// ---------------- residual + LayerNorm over 16384-element rows ----------------
__global__ void finalize_kernel(const float* __restrict__ recon,
                                const float* __restrict__ inputs,
                                float* __restrict__ out){
    int row = blockIdx.x;                         // b*16 + l
    size_t base = (size_t)row * (2 * TABLE_DIM);
    __shared__ float s1[256], s2[256];
    float sum = 0.0f, sq = 0.0f;
    for (int k = threadIdx.x; k < 2 * TABLE_DIM; k += 256){
        float v = recon[base + k] + inputs[base + k];
        sum += v; sq += v * v;
    }
    s1[threadIdx.x] = sum; s2[threadIdx.x] = sq;
    __syncthreads();
    for (int st = 128; st > 0; st >>= 1){
        if (threadIdx.x < st){
            s1[threadIdx.x] += s1[threadIdx.x + st];
            s2[threadIdx.x] += s2[threadIdx.x + st];
        }
        __syncthreads();
    }
    float mu  = s1[0] * (1.0f / (2 * TABLE_DIM));
    float var = s2[0] * (1.0f / (2 * TABLE_DIM)) - mu * mu;
    float rstd = rsqrtf(var + 1e-5f);
    for (int k = threadIdx.x; k < 2 * TABLE_DIM; k += 256){
        float v = recon[base + k] + inputs[base + k];
        out[base + k] = (v - mu) * rstd;
    }
}

extern "C" void kernel_launch(void* const* d_in, const int* in_sizes, int n_in,
                              void* d_out, int out_size, void* d_ws, size_t ws_size,
                              hipStream_t stream) {
    const float* inputs   = (const float*)d_in[0];   // [8,16,16384]
    const float* s        = (const float*)d_in[1];   // [8,512]
    // d_in[2] = coords (regular grid; recomputed analytically in-kernel)
    const float* affine_w = (const float*)d_in[3];   // [32,512]
    const float* affine_b = (const float*)d_in[4];   // [32]
    const float* conv_w   = (const float*)d_in[5];   // [32,32,3,3]
    const float* conv_b   = (const float*)d_in[6];   // [32]
    float* out = (float*)d_out;

    char* w = (char*)d_ws;
    float*     styles = (float*)w;                      w += BATCH * CH * sizeof(float);        // 1 KB
    float*     dcoef  = (float*)w;                      w += BATCH * CH * sizeof(float);        // 1 KB
    _Float16*  wt     = (_Float16*)w;                   w += 9 * CH * CH * sizeof(_Float16);    // 18 KB
    float*     recon  = (float*)w;                      w += (size_t)BATCH * TABLE_NUM * TABLE_DIM * 2 * sizeof(float); // 8 MB
    _Float16*  xm     = (_Float16*)w;                   // BATCH * 258*258 * 32 f16  (~34 MB)

    const int recon_elems = BATCH * TABLE_NUM * TABLE_DIM * 2;   // 2,097,152
    zero_kernel<<<(recon_elems + 255) / 256, 256, 0, stream>>>(recon, recon_elems);
    styles_kernel<<<BATCH, CH, 0, stream>>>(s, affine_w, affine_b, conv_w, styles, dcoef);
    wprep_kernel<<<(9 * CH * CH + 255) / 256, 256, 0, stream>>>(conv_w, wt);
    retrieve_kernel<<<dim3((PAD * PAD + 255) / 256, BATCH), 256, 0, stream>>>(inputs, styles, xm);
    conv_kernel<<<dim3(RESO / 64, RESO, BATCH), 128, 0, stream>>>(xm, wt, dcoef, conv_b, recon);
    finalize_kernel<<<BATCH * TABLE_NUM, 256, 0, stream>>>(recon, inputs, out);
}
